// Non_local_Attn_5677946765888
// MI455X (gfx1250) — matrix operand order repaired and checked by reading, not compiled, not run on hardware
//
#include <hip/hip_runtime.h>

#define B_   4
#define C_   256
#define N_   4096
#define G_   8
#define CPG_ 32
#define EPS_ 1e-5f

typedef __attribute__((ext_vector_type(16))) _Float16 v16h;
typedef __attribute__((ext_vector_type(8)))  float    v8f;

// ---------------- WMMA fragment helpers (contiguous k order, k = 16*(lane/16) + s) ----------------

// A-matrix 16x32 f16, source row-major [m][k] with row stride ld (elements).
// lanes 0-15: M=lane, K=0-15; lanes 16-31: M=lane-16, K=16-31
__device__ __forceinline__ v16h load_a_frag(const _Float16* __restrict__ base,
                                            int ld, int lane) {
  const int m   = lane & 15;
  const int khi = (lane >> 4) << 4;   // 0 or 16
  v16h a;
#pragma unroll
  for (int r = 0; r < 8; ++r) {
    const int k = (r << 1) + khi;
    const _Float16* p = base + (size_t)m * ld + k;
    a[2 * r]     = p[0];
    a[2 * r + 1] = p[1];
  }
  return a;
}

// B-matrix 32x16, where B[k][n] = src[n*ld + k] ("n-major, k contiguous").
// lanes 0-15: N=lane, K=0-15; lanes 16-31: N=lane-16, K=16-31.
__device__ __forceinline__ v16h load_b_frag_t(const _Float16* __restrict__ base,
                                              int ld, int lane) {
  const int n  = lane & 15;
  const int kb = (lane >> 4) << 4;    // 0 or 16
  v16h b;
#pragma unroll
  for (int r = 0; r < 8; ++r) {
    const _Float16* p = base + (size_t)n * ld + kb + 2 * r;
    b[2 * r]     = p[0];
    b[2 * r + 1] = p[1];
  }
  return b;
}

__device__ __forceinline__ v8f wmma16(v16h a, v16h b, v8f c) {
  return __builtin_amdgcn_wmma_f32_16x16x32_f16(
      false, a, false, b, (short)0, c, false, false);
}

// ---------------- CDNA5 async copy (global -> LDS, ASYNCcnt) ----------------
__device__ __forceinline__ void async_load_b128(unsigned lds_off, unsigned gvoff,
                                                const void* sbase) {
  // GVS form: mem_addr = SGPR_base + VGPR_off(i32); vdst = LDS byte address
  asm volatile("global_load_async_to_lds_b128 %0, %1, %2"
               :: "v"(lds_off), "v"(gvoff), "s"(sbase)
               : "memory");
}
__device__ __forceinline__ void wait_async_8() {
  asm volatile("s_wait_asynccnt 8" ::: "memory");
}
__device__ __forceinline__ void wait_async_0() {
  asm volatile("s_wait_asynccnt 0" ::: "memory");
}

// ---------------- Kernel 1: GroupNorm statistics ----------------
__global__ void gn_stats_kernel(const float* __restrict__ x,
                                float* __restrict__ stats) {
  __shared__ float red0[256];
  __shared__ float red1[256];
  const int bg = blockIdx.x;               // b*G + g
  const int b = bg / G_, g = bg % G_;
  const float* base = x + ((size_t)b * C_ + (size_t)g * CPG_) * N_;
  float sum = 0.f, sq = 0.f;
  for (int i = threadIdx.x; i < CPG_ * N_; i += blockDim.x) {
    const float v = base[i];
    sum += v;
    sq  += v * v;
  }
  red0[threadIdx.x] = sum;
  red1[threadIdx.x] = sq;
  __syncthreads();
  for (int st = 128; st > 0; st >>= 1) {
    if ((int)threadIdx.x < st) {
      red0[threadIdx.x] += red0[threadIdx.x + st];
      red1[threadIdx.x] += red1[threadIdx.x + st];
    }
    __syncthreads();
  }
  if (threadIdx.x == 0) {
    const float cnt = (float)(CPG_ * N_);
    const float mu  = red0[0] / cnt;
    const float var = red1[0] / cnt - mu * mu;
    stats[bg * 2]     = mu;
    stats[bg * 2 + 1] = rsqrtf(var + EPS_);
  }
}

// ------- Kernel 2: GroupNorm apply + LDS-tiled transpose to f16 [p][c] ------
__global__ __launch_bounds__(256) void gn_apply_kernel(
    const float* __restrict__ x, const float* __restrict__ gamma,
    const float* __restrict__ beta, const float* __restrict__ stats,
    _Float16* __restrict__ hT) {
  __shared__ float tile[32][33];
  const int b  = blockIdx.z;
  const int c0 = blockIdx.y * 32;          // one 32-channel tile == one group
  const int p0 = blockIdx.x * 32;

  // coalesced load: vary p fastest
  {
    const int tp = threadIdx.x & 31;
    const int tc = threadIdx.x >> 5;       // 0..7
#pragma unroll
    for (int i = 0; i < 4; ++i) {
      const int cc = tc + i * 8;
      tile[cc][tp] = x[((size_t)b * C_ + c0 + cc) * N_ + p0 + tp];
    }
  }
  __syncthreads();

  // coalesced store: vary c fastest
  const int wc = threadIdx.x & 31;
  const int wp = threadIdx.x >> 5;
  const int c = c0 + wc;
  const int g = c / CPG_;                  // uniform within block
  const float mu = stats[(b * G_ + g) * 2];
  const float rs = stats[(b * G_ + g) * 2 + 1];
  const float ga = gamma[c];
  const float be = beta[c];
#pragma unroll
  for (int i = 0; i < 4; ++i) {
    const int p = p0 + wp + i * 8;
    const float v = tile[wc][wp + i * 8];
    hT[((size_t)b * N_ + p) * C_ + c] = (_Float16)(((v - mu) * rs) * ga + be);
  }
}

// ---------------- Kernel 3: weights fp32 -> f16 ----------------
__global__ void wconv_kernel(const float* __restrict__ wq,
                             const float* __restrict__ wk,
                             const float* __restrict__ wv,
                             _Float16* __restrict__ wf) {
  const int idx = blockIdx.x * blockDim.x + threadIdx.x;
  const int n = C_ * C_;
  if (idx < n) {
    wf[idx]         = (_Float16)wq[idx];
    wf[n + idx]     = (_Float16)wk[idx];
    wf[2 * n + idx] = (_Float16)wv[idx];
  }
}

// ---------------- Kernel 4: QKV projections via WMMA ----------------
// q,k stored [p][c]; v stored [c][p] (so attention V-fragments are contiguous)
__global__ __launch_bounds__(32) void qkv_kernel(
    const _Float16* __restrict__ hT, const _Float16* __restrict__ wf,
    const float* __restrict__ bq, const float* __restrict__ bk,
    const float* __restrict__ bv,
    _Float16* __restrict__ qT, _Float16* __restrict__ kT,
    _Float16* __restrict__ vStd) {
  const int lane  = threadIdx.x;
  const int which = blockIdx.y;            // 0=q,1=k,2=v
  const int b     = blockIdx.z;
  const int tiles_n = C_ / 16;             // 16
  const int tm = blockIdx.x / tiles_n;
  const int tn = blockIdx.x % tiles_n;

  const _Float16* hb = hT + (size_t)b * N_ * C_;
  const _Float16* w  = wf + (size_t)which * C_ * C_;
  const float* bias  = (which == 0) ? bq : (which == 1) ? bk : bv;
  _Float16* outp = ((which == 0) ? qT : (which == 1) ? kT : vStd) +
                   (size_t)b * N_ * C_;

  v8f acc = {};
#pragma unroll
  for (int kk = 0; kk < 8; ++kk) {
    v16h a  = load_a_frag(hb + (size_t)tm * 16 * C_ + kk * 32, C_, lane);
    v16h bf = load_b_frag_t(w + (size_t)tn * 16 * C_ + kk * 32, C_, lane);
    acc = wmma16(a, bf, acc);
  }
  const int oc = tn * 16 + (lane & 15);
  const float bb = bias[oc];
  const int rbase = tm * 16 + ((lane >> 4) << 3);
  if (which == 2) {
#pragma unroll
    for (int v = 0; v < 8; ++v)
      outp[(size_t)oc * N_ + (rbase + v)] = (_Float16)(acc[v] + bb);
  } else {
#pragma unroll
    for (int v = 0; v < 8; ++v)
      outp[(size_t)(rbase + v) * C_ + oc] = (_Float16)(acc[v] + bb);
  }
}

// ---------------- Kernel 5: flash attention + residual ----------------
// Block = 4 waves, each wave owns 16 query rows. The 32-key V block is staged
// into LDS with async global->LDS copies (double-buffered, shared by 4 waves).
__global__ __launch_bounds__(128) void attn_kernel(
    const _Float16* __restrict__ qT, const _Float16* __restrict__ kT,
    const _Float16* __restrict__ vStd, const float* __restrict__ x,
    float* __restrict__ out) {
  __shared__ __align__(16) _Float16 sV[2][256 * 32];   // [c][j], 16 KB each
  __shared__ _Float16 ldsP[4][16 * 32];                // per-wave P staging

  const int tid  = threadIdx.x;
  const int lane = tid & 31;
  const int wid  = tid >> 5;
  const int b    = blockIdx.y;
  const int p0   = (blockIdx.x * 4 + wid) * 16;

  const _Float16* Q  = qT   + (size_t)b * N_ * C_;   // [p][c]
  const _Float16* Kb = kT   + (size_t)b * N_ * C_;   // [p][c]
  const _Float16* Vb = vStd + (size_t)b * N_ * C_;   // [c][p]

  // stage V columns [j0, j0+32) for all 256 channels into sV[buf] ([c][j])
  auto issue_v_stage = [&](int j0, int buf) {
    const unsigned vl0 = (unsigned)(uintptr_t)&sV[buf][0];
#pragma unroll
    for (int i = 0; i < 8; ++i) {
      const int c  = tid * 2 + (i >> 2);             // 0..255
      const int pc = i & 3;                          // 16B chunk within row
      const unsigned goff = ((unsigned)c * N_ + (unsigned)j0) * 2u + pc * 16u;
      const unsigned loff = (unsigned)c * 64u + pc * 16u;
      async_load_b128(vl0 + loff, goff, (const void*)Vb);
    }
  };

  // preload Q fragments for all 8 K-steps (reused for every key tile)
  v16h aq[8];
#pragma unroll
  for (int kk = 0; kk < 8; ++kk)
    aq[kk] = load_a_frag(Q + (size_t)p0 * C_ + kk * 32, C_, lane);

  float m[8], s[8];
  v8f o[16];
  const v8f vzero = {};
#pragma unroll
  for (int v = 0; v < 8; ++v) { m[v] = -__builtin_inff(); s[v] = 0.f; }
#pragma unroll
  for (int t = 0; t < 16; ++t) o[t] = vzero;

  const float scale = 0.0625f;             // 256^-0.5
  _Float16* lp = ldsP[wid];

  issue_v_stage(0, 0);

  for (int j0 = 0; j0 < N_; j0 += 32) {
    const int buf = (j0 >> 5) & 1;
    __syncthreads();                       // WAR: all waves done with buf^1
    const bool hasNext = (j0 + 32) < N_;
    if (hasNext) {
      issue_v_stage(j0 + 32, buf ^ 1);
      wait_async_8();                      // current stage complete
    } else {
      wait_async_0();
    }
    __syncthreads();                       // staged V visible to all waves

    // --- scores: S = Q x K^T for 32 keys (two 16x16 tiles) ---
    v8f sc0 = vzero, sc1 = vzero;
#pragma unroll
    for (int kk = 0; kk < 8; ++kk) {
      v16h bk0 = load_b_frag_t(Kb + (size_t)j0 * C_ + kk * 32, C_, lane);
      sc0 = wmma16(aq[kk], bk0, sc0);
      v16h bk1 = load_b_frag_t(Kb + (size_t)(j0 + 16) * C_ + kk * 32, C_, lane);
      sc1 = wmma16(aq[kk], bk1, sc1);
    }

    // --- online softmax update (per row, half-wave reductions) ---
    float pe0[8], pe1[8];
#pragma unroll
    for (int v = 0; v < 8; ++v) {
      const float a0 = sc0[v] * scale;
      const float a1 = sc1[v] * scale;
      float rm = fmaxf(a0, a1);
#pragma unroll
      for (int mk = 1; mk < 16; mk <<= 1)
        rm = fmaxf(rm, __shfl_xor(rm, mk, 32));
      const float nm = fmaxf(m[v], rm);
      const float alpha = __expf(m[v] - nm);
      const float e0 = __expf(a0 - nm);
      const float e1 = __expf(a1 - nm);
      float rs = e0 + e1;
#pragma unroll
      for (int mk = 1; mk < 16; mk <<= 1)
        rs += __shfl_xor(rs, mk, 32);
      s[v] = s[v] * alpha + rs;
      m[v] = nm;
      pe0[v] = e0;
      pe1[v] = e1;
#pragma unroll
      for (int t = 0; t < 16; ++t) o[t][v] *= alpha;
    }

    // --- stage P through LDS: C/D layout -> A-matrix layout ---
    {
      const int row = (lane >> 4) << 3;    // 0 or 8
      const int col = lane & 15;
#pragma unroll
      for (int v = 0; v < 8; ++v) {
        lp[(row + v) * 32 + col]      = (_Float16)pe0[v];
        lp[(row + v) * 32 + 16 + col] = (_Float16)pe1[v];
      }
    }
    v16h pA = load_a_frag(lp, 32, lane);

    // --- O += P x V from LDS (16 channel tiles, contiguous ds_load_b128) ---
#pragma unroll
    for (int t = 0; t < 16; ++t) {
      v16h bv = load_b_frag_t(&sV[buf][(size_t)t * 16 * 32], 32, lane);
      o[t] = wmma16(pA, bv, o[t]);
    }
  }

  // --- epilogue: out[b][c][p] = x + O / l ---
#pragma unroll
  for (int v = 0; v < 8; ++v) {
    const float inv = 1.0f / s[v];
    const int p = p0 + ((lane >> 4) << 3) + v;
#pragma unroll
    for (int t = 0; t < 16; ++t) {
      const int c = t * 16 + (lane & 15);
      const size_t idx = ((size_t)(b * C_ + c)) * N_ + p;
      out[idx] = x[idx] + o[t][v] * inv;
    }
  }
}

// ---------------- Host launcher ----------------
extern "C" void kernel_launch(void* const* d_in, const int* in_sizes, int n_in,
                              void* d_out, int out_size, void* d_ws,
                              size_t ws_size, hipStream_t stream) {
  (void)in_sizes; (void)n_in; (void)out_size; (void)ws_size;
  const float* x     = (const float*)d_in[0];
  const float* gamma = (const float*)d_in[1];
  const float* beta  = (const float*)d_in[2];
  const float* wq    = (const float*)d_in[3];
  const float* bq    = (const float*)d_in[4];
  const float* wk    = (const float*)d_in[5];
  const float* bk    = (const float*)d_in[6];
  const float* wv    = (const float*)d_in[7];
  const float* bv    = (const float*)d_in[8];
  float* out = (float*)d_out;

  char* ws = (char*)d_ws;
  size_t off = 0;
  float* stats  = (float*)(ws + off);     off += 256;                       // B*G*2
  _Float16* hT  = (_Float16*)(ws + off);  off += (size_t)B_ * N_ * C_ * 2;  // 8 MB
  _Float16* qT  = (_Float16*)(ws + off);  off += (size_t)B_ * N_ * C_ * 2;
  _Float16* kT  = (_Float16*)(ws + off);  off += (size_t)B_ * N_ * C_ * 2;
  _Float16* vSt = (_Float16*)(ws + off);  off += (size_t)B_ * N_ * C_ * 2;
  _Float16* wf  = (_Float16*)(ws + off);  off += (size_t)3 * C_ * C_ * 2;   // 384 KB

  gn_stats_kernel<<<B_ * G_, 256, 0, stream>>>(x, stats);
  gn_apply_kernel<<<dim3(N_ / 32, C_ / 32, B_), 256, 0, stream>>>(
      x, gamma, beta, stats, hT);
  wconv_kernel<<<(C_ * C_ + 255) / 256, 256, 0, stream>>>(wq, wk, wv, wf);
  qkv_kernel<<<dim3((N_ / 16) * (C_ / 16), 3, B_), 32, 0, stream>>>(
      hT, wf, bq, bk, bv, qT, kT, vSt);
  attn_kernel<<<dim3(N_ / 64, B_), 128, 0, stream>>>(qT, kT, vSt, x, out);
}
